// TripletLoss_44023414784277
// MI455X (gfx1250) — 16-bit and 8-bit matrix operand order repaired and checked by reading, compiled, not run on hardware
//
#include <hip/hip_runtime.h>


#define N    384
#define D    512
#define L    80
#define LPAD 128
#define NT   24

typedef __attribute__((ext_vector_type(16))) __bf16 v16bf;
typedef __attribute__((ext_vector_type(8)))  float  v8f;
typedef __attribute__((ext_vector_type(8)))  int    v8i;

#define WS_INV   0
#define WS_PSUM  1536
#define WS_PCNT  3072
#define WS_PK    4608
#define WS_SIMO  53760
#define WS_W     643584
#define WS_PD    1233408
#define WS_SHI   1823232
#define WS_SLO   2216448

__global__ void k_rownorm(const float* __restrict__ src, float* __restrict__ inv) {
  __shared__ float red[256];
  const int r = blockIdx.x, t = threadIdx.x;
  float a = src[r * D + t];
  float b = src[r * D + t + 256];
  red[t] = a * a + b * b;
  __syncthreads();
  for (int s = 128; s > 0; s >>= 1) {
    if (t < s) red[t] += red[t + s];
    __syncthreads();
  }
  if (t == 0) inv[r] = 1.0f / fmaxf(sqrtf(red[0]), 1e-8f);
}

__global__ void k_pack(const int* __restrict__ lab, unsigned char* __restrict__ pk) {
  const int idx = blockIdx.x * 256 + threadIdx.x;
  const int r = idx / LPAD, c = idx - r * LPAD;
  pk[idx] = (c < L) ? (unsigned char)lab[r * L + c] : (unsigned char)0;
}

__global__ void k_split(const float* __restrict__ src,
                        __bf16* __restrict__ shi, __bf16* __restrict__ slo) {
  const int idx = blockIdx.x * 256 + threadIdx.x;
  const float x = src[idx];
  const unsigned u = __builtin_bit_cast(unsigned, x) & 0xFFFF0000u;
  const float h = __builtin_bit_cast(float, u);
  shi[idx] = (__bf16)h;
  slo[idx] = (__bf16)(x - h);
}

__global__ void k_gram_labels(const unsigned char* __restrict__ pk,
                              int* __restrict__ simo) {
  const int wave = threadIdx.x >> 5, lane = threadIdx.x & 31;
  const int tile = blockIdx.x * 4 + wave;
  const int rt = (tile / NT) * 16, ct = (tile % NT) * 16;
  const int m = lane & 15, h = lane >> 4;

  v8i acc = {};
#pragma unroll
  for (int k0 = 0; k0 < LPAD; k0 += 64) {
    const unsigned char* arow = pk + (rt + m) * LPAD + k0;
    const unsigned char* brow = pk + (ct + m) * LPAD + k0;
    v8i a, b;
#pragma unroll
    for (int j = 0; j < 8; ++j) {
      a[j] = *(const int*)(arow + 16 * (j >> 1) + 8 * h + 4 * (j & 1));
      b[j] = *(const int*)(brow + 16 * (j >> 1) + 8 * h + 4 * (j & 1));
    }
    acc = __builtin_amdgcn_wmma_i32_16x16x64_iu8(false, a, false, b, acc,
                                                 false, false);
  }
#pragma unroll
  for (int r2 = 0; r2 < 8; ++r2)
    simo[(rt + r2 + 8 * h) * N + ct + m] = acc[r2];
}

__global__ void k_ndcg(const int* __restrict__ simo, float* __restrict__ w) {
  __shared__ int   hist[96];
  __shared__ float pref[N + 1];
  __shared__ float Zs;
  const int i = blockIdx.x, t = threadIdx.x;

  if (t < 96) hist[t] = 0;
  __syncthreads();

  int sv[3];
#pragma unroll
  for (int q = 0; q < 3; ++q) {
    const int j = t + q * 128;
    int s = simo[i * N + j];
    sv[q] = s;
    int b = s < 0 ? 0 : (s > 80 ? 80 : s);
    atomicAdd(&hist[b], 1);
  }
  for (int q = t; q <= N; q += 128)
    pref[q] = (q == 0) ? 0.0f : 1.0f / log2f((float)(q + 1));
  __syncthreads();

  if (t == 0) {
    float acc = 0.0f;
    for (int p = 1; p <= N; ++p) { acc += pref[p]; pref[p] = acc; }
    float Z = 0.0f;
    int pos = 0;
    for (int v = 80; v >= 1; --v) {
      const int c = hist[v];
      if (c) {
        Z += (exp2f((float)v) - 1.0f) * (pref[pos + c] - pref[pos]);
        pos += c;
      }
    }
    Zs = Z;
  }
  __syncthreads();

  const float invZ = 1.0f / Zs;
#pragma unroll
  for (int q = 0; q < 3; ++q) {
    const int j = t + q * 128;
    w[i * N + j] = (exp2f((float)sv[q]) - 1.0f) * invZ;
  }
}

__global__ void k_gram_src(const __bf16* __restrict__ shi,
                           const __bf16* __restrict__ slo,
                           const float* __restrict__ inv,
                           float* __restrict__ pd) {
  const int wave = threadIdx.x >> 5, lane = threadIdx.x & 31;
  const int tile = blockIdx.x * 4 + wave;
  const int rt = (tile / NT) * 16, ct = (tile % NT) * 16;
  const int m = lane & 15, h = lane >> 4;

  const __bf16* ahr = shi + (rt + m) * D;
  const __bf16* alr = slo + (rt + m) * D;
  const __bf16* bhr = shi + (ct + m) * D;
  const __bf16* blr = slo + (ct + m) * D;

  v8f c = {};
#pragma unroll 4
  for (int k0 = 0; k0 < D; k0 += 32) {
    v16bf ahi, alo;
    ((uint4*)&ahi)[0] = *(const uint4*)(ahr + k0 + 8 * h);
    ((uint4*)&ahi)[1] = *(const uint4*)(ahr + k0 + 16 + 8 * h);
    ((uint4*)&alo)[0] = *(const uint4*)(alr + k0 + 8 * h);
    ((uint4*)&alo)[1] = *(const uint4*)(alr + k0 + 16 + 8 * h);
    v16bf bhi, blo;
    ((uint4*)&bhi)[0] = *(const uint4*)(bhr + k0 + 8 * h);
    ((uint4*)&bhi)[1] = *(const uint4*)(bhr + k0 + 16 + 8 * h);
    ((uint4*)&blo)[0] = *(const uint4*)(blr + k0 + 8 * h);
    ((uint4*)&blo)[1] = *(const uint4*)(blr + k0 + 16 + 8 * h);

    c = __builtin_amdgcn_wmma_f32_16x16x32_bf16(false, ahi, false, bhi,
                                                (short)0, c, false, false);
    c = __builtin_amdgcn_wmma_f32_16x16x32_bf16(false, ahi, false, blo,
                                                (short)0, c, false, false);
    c = __builtin_amdgcn_wmma_f32_16x16x32_bf16(false, alo, false, bhi,
                                                (short)0, c, false, false);
  }

  const float invc = inv[ct + m];
#pragma unroll
  for (int r2 = 0; r2 < 8; ++r2) {
    const int row = rt + r2 + 8 * h;
    const float g = c[r2] * inv[row] * invc;
    pd[row * N + ct + m] = fmaxf(1.0f - g, 0.0f);
  }
}

__global__ void k_triplet(const float* __restrict__ w,
                          const float* __restrict__ pd,
                          const int* __restrict__ simo,
                          float* __restrict__ psum, unsigned* __restrict__ pcnt) {
  __shared__ float sw[N], sp[N];
  __shared__ unsigned char sf[N];
  __shared__ float rs[256];
  __shared__ unsigned rc[256];
  const int i = blockIdx.x, t = threadIdx.x;

  for (int j = t; j < N; j += 256) {
    sw[j] = w[i * N + j];
    sp[j] = pd[i * N + j];
    sf[j] = (unsigned char)(simo[i * N + j] > 0);
  }
  __syncthreads();

  float sum = 0.0f;
  unsigned cnt = 0;
  for (int j = 0; j < N; ++j) {
    if (!sf[j]) continue;
    const float wj = sw[j], pj = sp[j];
    for (int k = t; k < N; k += 256) {
      if (!sf[k]) {
        const float term = 10.0f * (wj - sw[k]) * (pj - sp[k]);
        if (term > 0.0f)   sum += term;
        if (term > 1e-16f) cnt++;
      }
    }
  }
  rs[t] = sum;
  rc[t] = cnt;
  __syncthreads();
  for (int s = 128; s > 0; s >>= 1) {
    if (t < s) { rs[t] += rs[t + s]; rc[t] += rc[t + s]; }
    __syncthreads();
  }
  if (t == 0) { psum[i] = rs[0]; pcnt[i] = rc[0]; }
}

__global__ void k_final(const float* __restrict__ psum,
                        const unsigned* __restrict__ pcnt,
                        float* __restrict__ out) {
  __shared__ float rs[128];
  __shared__ unsigned rc[128];
  const int t = threadIdx.x;
  rs[t] = psum[t] + psum[t + 128] + psum[t + 256];
  rc[t] = pcnt[t] + pcnt[t + 128] + pcnt[t + 256];
  __syncthreads();
  for (int s = 64; s > 0; s >>= 1) {
    if (t < s) { rs[t] += rs[t + s]; rc[t] += rc[t + s]; }
    __syncthreads();
  }
  if (t == 0) out[0] = rs[0] / ((float)rc[0] + 1e-16f);
}

extern "C" void kernel_launch(void* const* d_in, const int* in_sizes, int n_in,
                              void* d_out, int out_size, void* d_ws, size_t ws_size,
                              hipStream_t stream) {
  const float* src = (const float*)d_in[0];
  const int*   lab = (const int*)d_in[1];

  char* ws = (char*)d_ws;
  float*         inv  = (float*)(ws + WS_INV);
  float*         psum = (float*)(ws + WS_PSUM);
  unsigned*      pcnt = (unsigned*)(ws + WS_PCNT);
  unsigned char* pk   = (unsigned char*)(ws + WS_PK);
  int*           simo = (int*)(ws + WS_SIMO);
  float*         wbuf = (float*)(ws + WS_W);
  float*         pdb  = (float*)(ws + WS_PD);
  __bf16*        shi  = (__bf16*)(ws + WS_SHI);
  __bf16*        slo  = (__bf16*)(ws + WS_SLO);
  float*         out  = (float*)d_out;

  k_rownorm    <<<N, 256, 0, stream>>>(src, inv);
  k_pack       <<<(N * LPAD) / 256, 256, 0, stream>>>(lab, pk);
  k_split      <<<(N * D) / 256, 256, 0, stream>>>(src, shi, slo);
  k_gram_labels<<<(NT * NT) / 4, 128, 0, stream>>>(pk, simo);
  k_ndcg       <<<N, 128, 0, stream>>>(simo, wbuf);
  k_gram_src   <<<(NT * NT) / 4, 128, 0, stream>>>(shi, slo, inv, pdb);
  k_triplet    <<<N, 256, 0, stream>>>(wbuf, pdb, simo, psum, pcnt);
  k_final      <<<1, 128, 0, stream>>>(psum, pcnt, out);
}
